// DKD_18459769438577
// MI455X (gfx1250) — matrix operand order repaired and checked by reading, not compiled, not run on hardware
//
#include <hip/hip_runtime.h>

// DKD forward for MI455X (gfx1250, wave32).
// Pipeline:
//   K1: zero counters/histogram.
//   K2: fused 2-iteration simple_nms (5 chained 5x5 max-pools) in LDS,
//       32x32 tile + halo 10 (52x52), emits NMS-surviving candidates +
//       1024-bin score histogram (scores ~ U(0,1)).
//   K3: histogram threshold -> select 4096 keypoints per image.
//   K4: per-keypoint 5x5 soft-argmax via V_WMMA_F32_16X16X32_F16:
//       x_exp(16x32) x G(32x16) with G columns = [dx, dy, 1, dx^2+dy^2],
//       then subpixel refine + bilinear resample + dispersity.

#define IW 1024
#define IH 1024
#define NB 8
#define TOPK 4096
#define CAP 131072   // per-image candidate capacity (NMS survivors ~40-80K)

typedef __attribute__((ext_vector_type(16))) _Float16 v16h;
typedef __attribute__((ext_vector_type(8)))  float    v8f;

__device__ __forceinline__ float max5(const float* a, int stride, int x, int y) {
  float m = -3e38f;
#pragma unroll
  for (int j = 0; j < 5; ++j)
#pragma unroll
    for (int i = 0; i < 5; ++i)
      m = fmaxf(m, a[(y + j) * stride + (x + i)]);
  return m;
}

__global__ void dkd_init_kernel(int* __restrict__ cnt, int* __restrict__ hist) {
  int i = blockIdx.x * blockDim.x + threadIdx.x;
  if (i < NB) cnt[i] = 0;
  if (i < NB * 1024) hist[i] = 0;
}

__global__ __launch_bounds__(256) void dkd_nms_kernel(
    const float* __restrict__ scores, int* __restrict__ cnt, int* __restrict__ hist,
    float* __restrict__ candS, int* __restrict__ candI) {
  // Region halos: s:10(52) -> mask0:8(48) -> supp1/sp1:6(44) -> mask1:4(40)
  //               -> supp2/sp2:2(36) -> out:0(32)
  __shared__ float sS [52 * 52];
  __shared__ float m0 [48 * 48];
  __shared__ float sp1[44 * 44];
  __shared__ float su1[44 * 44];
  __shared__ float m1 [40 * 40];
  __shared__ float sp2[36 * 36];
  __shared__ float su2[36 * 36];

  const int tid = threadIdx.x;
  const int b   = blockIdx.z;
  const int gx0 = blockIdx.x * 32, gy0 = blockIdx.y * 32;
  const float* img = scores + (size_t)b * (IW * IH);

  for (int i = tid; i < 52 * 52; i += 256) {
    int x = i % 52, y = i / 52;
    int gx = gx0 - 10 + x, gy = gy0 - 10 + y;
    bool v = ((unsigned)gx < IW) && ((unsigned)gy < IH);
    sS[i] = v ? img[gy * IW + gx] : -3e38f;   // -inf padding semantics
  }
  __syncthreads();

  // max_mask = (s == maxpool(s)), forced 0 outside image
  for (int i = tid; i < 48 * 48; i += 256) {
    int x = i % 48, y = i / 48;
    int gx = gx0 - 8 + x, gy = gy0 - 8 + y;
    bool v = ((unsigned)gx < IW) && ((unsigned)gy < IH);
    float c = sS[(y + 2) * 52 + x + 2];
    m0[i] = (v && c == max5(sS, 52, x, y)) ? 1.f : 0.f;
  }
  __syncthreads();

  // iter 1: supp = maxpool(mask)>0 ; supp_scores = where(supp, 0, s)
  for (int i = tid; i < 44 * 44; i += 256) {
    int x = i % 44, y = i / 44;
    bool supp = max5(m0, 48, x, y) > 0.f;
    su1[i] = supp ? 1.f : 0.f;
    sp1[i] = supp ? 0.f : sS[(y + 4) * 52 + x + 4];
  }
  __syncthreads();

  // mask1 = mask0 | (supp_scores==maxpool(supp_scores) & ~supp)
  for (int i = tid; i < 40 * 40; i += 256) {
    int x = i % 40, y = i / 40;
    float c = sp1[(y + 2) * 44 + x + 2];
    bool nm = (c == max5(sp1, 44, x, y)) && (su1[(y + 2) * 44 + x + 2] == 0.f);
    m1[i] = ((m0[(y + 4) * 48 + x + 4] != 0.f) || nm) ? 1.f : 0.f;
  }
  __syncthreads();

  // iter 2
  for (int i = tid; i < 36 * 36; i += 256) {
    int x = i % 36, y = i / 36;
    bool supp = max5(m1, 40, x, y) > 0.f;
    su2[i] = supp ? 1.f : 0.f;
    sp2[i] = supp ? 0.f : sS[(y + 8) * 52 + x + 8];
  }
  __syncthreads();

  for (int i = tid; i < 32 * 32; i += 256) {
    int x = i % 32, y = i / 32;
    float c = sp2[(y + 2) * 36 + x + 2];
    bool nm2 = (c == max5(sp2, 36, x, y)) && (su2[(y + 2) * 36 + x + 2] == 0.f);
    bool m2  = (m1[(y + 4) * 40 + x + 4] != 0.f) || nm2;
    int gx = gx0 + x, gy = gy0 + y;
    if (m2 && gx >= 2 && gx < IW - 2 && gy >= 2 && gy < IH - 2) {
      float sc = sS[(y + 10) * 52 + x + 10];
      if (sc > 0.f) {
        int bin = min(1023, (int)(sc * 1024.f));
        atomicAdd(&hist[b * 1024 + bin], 1);
        int pos = atomicAdd(&cnt[b], 1);
        if (pos < CAP) {
          candS[(size_t)b * CAP + pos] = sc;
          candI[(size_t)b * CAP + pos] = gy * IW + gx;
        }
      }
    }
  }
}

__global__ __launch_bounds__(256) void dkd_select_kernel(
    const int* __restrict__ cnt, const int* __restrict__ hist,
    const float* __restrict__ candS, const int* __restrict__ candI,
    int* __restrict__ sel) {
  __shared__ int sh_t;
  __shared__ int sh_n;
  const int b = blockIdx.x;
  const int tid = threadIdx.x;
  if (tid == 0) {
    int cum = 0, t = 0;
    for (int bin = 1023; bin >= 0; --bin) {
      cum += hist[b * 1024 + bin];
      if (cum >= TOPK) { t = bin; break; }
    }
    sh_t = t;
    sh_n = 0;
  }
  __syncthreads();
  const int t = sh_t;
  const int n = min(cnt[b], CAP);
  for (int i = tid; i < n; i += 256) {
    float s = candS[(size_t)b * CAP + i];
    int bin = min(1023, (int)(s * 1024.f));
    if (bin >= t) {
      int slot = atomicAdd(&sh_n, 1);
      if (slot < TOPK) sel[b * TOPK + slot] = candI[(size_t)b * CAP + i];
    }
  }
  __syncthreads();
  int filled = min(sh_n, TOPK);
  for (int s = filled + tid; s < TOPK; s += 256)
    sel[b * TOPK + s] = s;  // pad with low-index (zero-NMS) pixels like top_k over zeros
}

__global__ __launch_bounds__(256) void dkd_refine_kernel(
    const float* __restrict__ scores, const int* __restrict__ sel,
    float* __restrict__ out) {
  __shared__ _Float16 X[8][32][32];  // per-wave: 32 keypoints x 32 taps (25 used)
  __shared__ float    S[8][32][4];   // per-wave: [Sdx, Sdy, Ssum, Sd2]

  const int tid  = threadIdx.x;
  const int wave = tid >> 5, lane = tid & 31;
  const int b  = blockIdx.x >> 4;                 // 16 blocks per batch image
  const int kp = (blockIdx.x & 15) * 256 + tid;   // keypoint index within batch
  const float* img = scores + (size_t)b * (IW * IH);

  const int idx = sel[b * TOPK + kp];
  const int ky = idx >> 10, kx = idx & (IW - 1);

  // Gather 5x5 patch, compute max, store exp((v-max)/0.1) as f16
  float val[25];
  float mx = -3e38f;
#pragma unroll
  for (int p = 0; p < 25; ++p) {
    int dy = p / 5 - 2, dx = p % 5 - 2;
    int ny = min(max(ky + dy, 0), IH - 1);
    int nx = min(max(kx + dx, 0), IW - 1);
    float v = img[ny * IW + nx];
    val[p] = v;
    mx = fmaxf(mx, v);
  }
#pragma unroll
  for (int p = 0; p < 25; ++p)
    X[wave][lane][p] = (_Float16)__expf((val[p] - mx) * 10.f);
#pragma unroll
  for (int p = 25; p < 32; ++p) X[wave][lane][p] = (_Float16)0.f;
  __syncthreads();

  // Assemble WMMA fragments (16-bit A-matrix 16x32 layout, wave32)
  const int row = lane & 15, hi = lane >> 4;
  v16h a0, a1, bf;
#pragma unroll
  for (int e = 0; e < 16; ++e) {
    int k = e + 8 * ((e >> 3) + hi);       // A-matrix K index for this lane/elem
    a0[e] = X[wave][row][k];
    a1[e] = X[wave][16 + row][k];
    int kb = e + 8 * ((e >> 3) + hi);      // B-matrix K index
    float g = 0.f;
    if (kb < 25 && row < 4) {
      float dx = (float)(kb % 5 - 2), dy = (float)(kb / 5 - 2);
      g = (row == 0) ? dx : (row == 1) ? dy : (row == 2) ? 1.f : (dx * dx + dy * dy);
    }
    bf[e] = (_Float16)g;
  }
  v8f c = {};
  v8f d0 = __builtin_amdgcn_wmma_f32_16x16x32_f16(false, a0, false, bf, (short)0, c, false, false);
  v8f d1 = __builtin_amdgcn_wmma_f32_16x16x32_f16(false, a1, false, bf, (short)0, c, false, false);

  // Scatter the 4 useful columns of D to LDS (D elem r: M=r+8*hi, N=lane&15)
  if (row < 4) {
#pragma unroll
    for (int r = 0; r < 8; ++r) {
      S[wave][r + 8 * hi][row]      = d0[r];
      S[wave][16 + r + 8 * hi][row] = d1[r];
    }
  }
  __syncthreads();

  const float Sdx  = S[wave][lane][0];
  const float Sdy  = S[wave][lane][1];
  const float Ssum = S[wave][lane][2];
  const float Sd2  = S[wave][lane][3];
  const float inv  = 1.f / Ssum;
  const float rx = Sdx * inv, ry = Sdy * inv;
  // dispersity = sum xexp * ||(g - r)/R||^2 / Ssum, expanded (R=2)
  const float disp = (Sd2 - 2.f * (rx * Sdx + ry * Sdy) + (rx * rx + ry * ry) * Ssum) * inv * 0.25f;

  const float px = (float)kx + rx, py = (float)ky + ry;
  // bilinear sample, align_corners=True (normalize->denormalize round-trips)
  float fx = floorf(px), fy = floorf(py);
  float wx = px - fx, wy = py - fy;
  int x0 = min(max((int)fx, 0), IW - 1), y0 = min(max((int)fy, 0), IH - 1);
  int x1 = min(x0 + 1, IW - 1),          y1 = min(y0 + 1, IH - 1);
  float v00 = img[y0 * IW + x0], v01 = img[y0 * IW + x1];
  float v10 = img[y1 * IW + x0], v11 = img[y1 * IW + x1];
  float ks = v00 * (1.f - wx) * (1.f - wy) + v01 * wx * (1.f - wy)
           + v10 * (1.f - wx) * wy         + v11 * wx * wy;

  const size_t o = (size_t)b * TOPK + kp;
  out[2 * o]     = px * (2.f / (IW - 1)) - 1.f;  // kp_norm.x
  out[2 * o + 1] = py * (2.f / (IH - 1)) - 1.f;  // kp_norm.y
  out[(size_t)NB * TOPK * 2 + o] = ks;           // kptscores
  out[(size_t)NB * TOPK * 3 + o] = disp;         // scoredispersity
}

extern "C" void kernel_launch(void* const* d_in, const int* in_sizes, int n_in,
                              void* d_out, int out_size, void* d_ws, size_t ws_size,
                              hipStream_t stream) {
  const float* scores = (const float*)d_in[0];
  float* out = (float*)d_out;

  // Workspace layout (4-byte units): cnt[8] | hist[8*1024] | sel[8*4096] | candS | candI
  int* ws    = (int*)d_ws;
  int* cnt   = ws;
  int* hist  = ws + 8;
  int* sel   = ws + 8 + NB * 1024;
  float* candS = (float*)(ws + 8 + NB * 1024 + NB * TOPK);
  int*   candI = (int*)(candS + (size_t)NB * CAP);

  dkd_init_kernel<<<32, 256, 0, stream>>>(cnt, hist);
  dim3 g(IW / 32, IH / 32, NB);
  dkd_nms_kernel<<<g, 256, 0, stream>>>(scores, cnt, hist, candS, candI);
  dkd_select_kernel<<<NB, 256, 0, stream>>>(cnt, hist, candS, candI, sel);
  dkd_refine_kernel<<<NB * 16, 256, 0, stream>>>(scores, sel, out);
}
